// GatedSkipFusion_11991548690668
// MI455X (gfx1250) — matrix operand order repaired and checked by reading, not compiled, not run on hardware
//
#include <hip/hip_runtime.h>
#include <hip/hip_bf16.h>

typedef __attribute__((ext_vector_type(16))) _Float16 v16h;
typedef __attribute__((ext_vector_type(8)))  float    v8f;

#define TT 4
#define BB 8
#define CC 64
#define HW 12544                 // 112*112
#define QTOT (BB * HW)           // 100352 positions per time step
#define NTILES (QTOT / 16)       // 6272 (exact)
#define NELEM (TT * BB * CC * HW)  // 25,690,112
#define NSTAT (TT * BB * HW)       // 401,408 per channel

// Fast sigmoid: v_exp_f32 + v_add + v_rcp_f32 (no IEEE div sequence).
static __device__ __forceinline__ float fast_sigmoid(float x) {
#if __has_builtin(__builtin_amdgcn_rcpf)
  return __builtin_amdgcn_rcpf(1.0f + __expf(-x));
#else
  return 1.0f / (1.0f + __expf(-x));
#endif
}

// ---------------------------------------------------------------------------
// Kernel 1: fused 1x1 gating conv (WMMA) + sigmoid blend + 4-step LIF scan.
// Each wave: one 16-position tile, all 64 out-channels, all 4 time steps.
// WMMA D[c,16 x pos,16] = sum_kk A(Wg f16, 16x32) x B(combined f16, 32x16).
// Writes spikes (0/1 bytes) to workspace.
// ---------------------------------------------------------------------------
__global__ __launch_bounds__(256) void fuse_lif_kernel(
    const float* __restrict__ dec, const float* __restrict__ enc,
    const float* __restrict__ Wg,  const float* __restrict__ bg,
    unsigned char* __restrict__ spikes)
{
  const int lane = threadIdx.x & 31;
  const int wave = threadIdx.x >> 5;
  const int col  = lane & 15;   // WMMA N / M-row index owned by this lane
  const int kh   = lane >> 4;   // lane-half (selects K sub-group per ISA layout)

  // --- Stage all 16 A fragments (Wg as f16) into LDS in WMMA register order.
  // frag f = ct*4 + kk ; per lane 16 halves at [(f*32+lane)*16].
  __shared__ __align__(32) _Float16 aLds[16 * 32 * 16];   // 16 KB
  for (int f = wave; f < 16; f += 8) {
    const int ct = f >> 2, kk = f & 3;
    const int crow = ct * 16 + col;              // A row M = lane%16
    #pragma unroll
    for (int h = 0; h < 16; ++h) {
      const int K = h + 16 * kh;   // contiguous K order, matches B fill
      aLds[(f * 32 + lane) * 16 + h] = (_Float16)Wg[crow * 128 + 32 * kk + K];
    }
  }
  __syncthreads();

  const int tile = blockIdx.x * 8 + wave;        // grid chosen so tile < NTILES
  const int q0   = tile * 16;
  const int b    = q0 / HW;
  const int p    = (q0 - b * HW) + col;          // this lane's spatial position

  // Per-lane bias values for its 32 channels: c = ct*16 + r + 8*kh
  float bgv[4][8];
  #pragma unroll
  for (int ct = 0; ct < 4; ++ct)
    #pragma unroll
    for (int r = 0; r < 8; ++r)
      bgv[ct][r] = bg[ct * 16 + r + 8 * kh];

  // LIF membrane state for the 32 channels this lane owns.
  float v[4][8];
  #pragma unroll
  for (int ct = 0; ct < 4; ++ct)
    #pragma unroll
    for (int r = 0; r < 8; ++r) v[ct][r] = 0.0f;

  for (int t = 0; t < TT; ++t) {
    const int base = ((t * BB + b) * CC) * HW;   // (t,b) slab base, channel-major

    // Prefetch a few planes of the next time step while WMMAs run.
    if (t < 3) {
      const int nbase = (((t + 1) * BB + b) * CC) * HW;
      __builtin_prefetch(dec + nbase + (16 * kh) * HW + p, 0, 1);
      __builtin_prefetch(dec + nbase + (32 + 16 * kh) * HW + p, 0, 1);
      __builtin_prefetch(enc + nbase + (16 * kh) * HW + p, 0, 1);
      __builtin_prefetch(enc + nbase + (32 + 16 * kh) * HW + p, 0, 1);
    }

    // --- Load B fragments: combined[k, pos] as f16 (contiguous K order, as in A:
    // lane n holds column n; lane-half selects K 0..15 vs 16..31).
    v16h bf[4];
    #pragma unroll
    for (int kk = 0; kk < 4; ++kk) {
      const float* __restrict__ src = (kk < 2) ? dec : enc;
      const int cbase = 32 * (kk & 1) + 16 * kh;
      #pragma unroll
      for (int h = 0; h < 16; ++h)
        bf[kk][h] = (_Float16)src[base + (cbase + h) * HW + p];
    }

    // --- GEMM + epilogue per 16-channel tile.
    #pragma unroll
    for (int ct = 0; ct < 4; ++ct) {
      v8f acc;
      #pragma unroll
      for (int r = 0; r < 8; ++r) acc[r] = bgv[ct][r];   // start from bias
      #pragma unroll
      for (int kk = 0; kk < 4; ++kk) {
        const v16h af =
            *reinterpret_cast<const v16h*>(&aLds[((ct * 4 + kk) * 32 + lane) * 16]);
        acc = __builtin_amdgcn_wmma_f32_16x16x32_f16(
            false, af, false, bf[kk], (short)0, acc, false, false);
      }
      // D layout: lane owns pos=p, rows M = r + 8*kh -> channel ct*16 + M.
      #pragma unroll
      for (int r = 0; r < 8; ++r) {
        const int c   = ct * 16 + r + 8 * kh;
        const int idx = base + c * HW + p;
        const float g = fast_sigmoid(acc[r]);
        const float d = dec[idx];
        const float e = enc[idx];
        const float x = fmaf(g, d - e, e);       // g*d + (1-g)*e
        float vv = fmaf(v[ct][r], 0.5f, x);      // tau=2: v - v/tau + x
        const bool sp = (vv >= 0.15f);
        v[ct][r] = sp ? 0.0f : vv;               // hard reset
        spikes[idx] = sp ? (unsigned char)1 : (unsigned char)0;
      }
    }
  }
}

// ---------------------------------------------------------------------------
// Kernel 2: per-channel spike count -> BN scale/bias.
// spikes in {0,1}  =>  var = m - m^2, so only the count is needed.
// One block per channel.
// ---------------------------------------------------------------------------
__global__ __launch_bounds__(256) void stats_kernel(
    const unsigned char* __restrict__ spikes,
    const float* __restrict__ gamma, const float* __restrict__ beta,
    float* __restrict__ scale, float* __restrict__ bias)
{
  const int c = blockIdx.x;
  const int tid = threadIdx.x;
  unsigned int cnt = 0;
  for (int tb = 0; tb < TT * BB; ++tb) {
    const unsigned int* seg =
        (const unsigned int*)(spikes + (tb * CC + c) * HW);
    for (int i = tid; i < HW / 4; i += 256) {
      const unsigned int w = seg[i];
      cnt += (w & 0xFF) + ((w >> 8) & 0xFF) + ((w >> 16) & 0xFF) + (w >> 24);
    }
  }
  __shared__ unsigned int red[256];
  red[tid] = cnt;
  __syncthreads();
  for (int s = 128; s > 0; s >>= 1) {
    if (tid < s) red[tid] += red[tid + s];
    __syncthreads();
  }
  if (tid == 0) {
    const float m   = (float)red[0] / (float)NSTAT;
    const float var = m - m * m;                  // E[s^2] = E[s]
    const float sc  = gamma[c] * rsqrtf(var + 1e-5f);
    scale[c] = sc;
    bias[c]  = beta[c] - m * sc;
  }
}

// ---------------------------------------------------------------------------
// Kernel 3: out = spike ? scale[c]+bias[c] : bias[c].  4 bytes -> float4.
// ---------------------------------------------------------------------------
__global__ __launch_bounds__(256) void apply_kernel(
    const unsigned char* __restrict__ spikes,
    const float* __restrict__ scale, const float* __restrict__ bias,
    float* __restrict__ out)
{
  const int i4 = blockIdx.x * 256 + threadIdx.x;   // grid sized exactly
  const int i  = i4 * 4;
  const int c  = (i / HW) & (CC - 1);
  const unsigned int w = *(const unsigned int*)(spikes + i);
  const float sc = scale[c];
  const float bi = bias[c];
  float4 o;
  o.x = (w & 0x000000FFu) ? sc + bi : bi;
  o.y = (w & 0x0000FF00u) ? sc + bi : bi;
  o.z = (w & 0x00FF0000u) ? sc + bi : bi;
  o.w = (w & 0xFF000000u) ? sc + bi : bi;
  *reinterpret_cast<float4*>(out + i) = o;
}

// ---------------------------------------------------------------------------
extern "C" void kernel_launch(void* const* d_in, const int* in_sizes, int n_in,
                              void* d_out, int out_size, void* d_ws, size_t ws_size,
                              hipStream_t stream) {
  const float* dec   = (const float*)d_in[0];
  const float* enc   = (const float*)d_in[1];
  const float* Wg    = (const float*)d_in[2];
  const float* bg    = (const float*)d_in[3];
  const float* gamma = (const float*)d_in[4];
  const float* beta  = (const float*)d_in[5];
  float* out = (float*)d_out;

  unsigned char* spikes = (unsigned char*)d_ws;            // NELEM bytes
  float* scale = (float*)((char*)d_ws + (size_t)NELEM);    // 64 floats
  float* bias  = scale + CC;                               // 64 floats

  // 6272 tiles / 8 waves per block
  fuse_lif_kernel<<<NTILES / 8, 256, 0, stream>>>(dec, enc, Wg, bg, spikes);
  stats_kernel<<<CC, 256, 0, stream>>>(spikes, gamma, beta, scale, bias);
  apply_kernel<<<NELEM / 4 / 256, 256, 0, stream>>>(spikes, scale, bias, out);
}
